// RetentiveAttention_48576080118368
// MI455X (gfx1250) — matrix operand order repaired and checked by reading, not compiled, not run on hardware
//
#include <hip/hip_runtime.h>
#include <hip/hip_bf16.h>

// ---------------------------------------------------------------------------
// RetNet retention forward for MI455X (gfx1250, wave32, WMMA bf16 16x16x32).
// ~69 GFLOP vs ~100MB traffic (~700 FLOP/B) -> compute-bound: everything
// GEMM-shaped runs on v_wmma_f32_16x16x32_bf16. Async global->LDS staging
// (ASYNCcnt) used where tiles are already bf16; global_prefetch_b8 for the
// streamed K blocks.
// ---------------------------------------------------------------------------

typedef __attribute__((ext_vector_type(8)))  __bf16 v8bf;
typedef __attribute__((ext_vector_type(16))) __bf16 v16bf;
typedef __attribute__((ext_vector_type(8)))  float  v8f;
typedef __attribute__((ext_vector_type(4)))  int    v4i;

union V16U { v16bf v; v8bf h[2]; };
union U4U  { uint4 u; __bf16 b[8]; };

static constexpr int Bn = 2, Tn = 2048, Cn = 1024, Hn = 16, Dhn = 64;
static constexpr int Mn = Bn * Tn;                 // 4096 rows
static constexpr float EPSf = 1e-5f;

// ---- async global->LDS staging (CDNA5 GLOBAL_LOAD_ASYNC_TO_LDS_B128) ------
#if __has_builtin(__builtin_amdgcn_global_load_async_to_lds_b128)
#define HAVE_ASYNC_COPY 1
#endif

__device__ __forceinline__ void stage_b128(void* lds_dst, const void* gsrc) {
#ifdef HAVE_ASYNC_COPY
  __builtin_amdgcn_global_load_async_to_lds_b128(
      (__attribute__((address_space(1))) v4i*)(v4i*)gsrc,
      (__attribute__((address_space(3))) v4i*)lds_dst, 0, 0);
#else
  *(uint4*)lds_dst = *(const uint4*)gsrc;
#endif
}

__device__ __forceinline__ void stage_wait() {
#ifdef HAVE_ASYNC_COPY
#if __has_builtin(__builtin_amdgcn_s_wait_asynccnt)
  __builtin_amdgcn_s_wait_asynccnt(0);
#else
  asm volatile("s_wait_asynccnt 0x0" ::: "memory");
#endif
#endif
}

__device__ __forceinline__ v8f wmma_bf16(v16bf a, v16bf b, v8f c) {
  return __builtin_amdgcn_wmma_f32_16x16x32_bf16(false, a, false, b,
                                                 (short)0, c, false, false);
}

__device__ __forceinline__ v8f vzero() {
  return v8f{0.f, 0.f, 0.f, 0.f, 0.f, 0.f, 0.f, 0.f};
}

// A-frag (16x32 bf16): lane<16 row=lane, K 0..15; lane>=16 K 16..31
__device__ __forceinline__ v16bf load_a_frag(const __bf16* base, int ldk,
                                             int row, int koff, int lane) {
  int half = lane >> 4;
  const __bf16* p = base + (row + (lane & 15)) * ldk + koff + half * 16;
  V16U u;
  u.h[0] = *(const v8bf*)(p);
  u.h[1] = *(const v8bf*)(p + 8);
  return u.v;
}

// B-frag (32x16 bf16) from LDS stored [n][k]: lane<16 col=lane K 0..15,
// lane>=16 col=lane-16 K 16..31 -> contiguous 16-element run.
__device__ __forceinline__ v16bf load_b_frag(const __bf16* base, int ldk,
                                             int col, int koff, int lane) {
  int half = lane >> 4;
  const __bf16* p = base + (col + (lane & 15)) * ldk + koff + half * 16;
  V16U u;
  u.h[0] = *(const v8bf*)(p);
  u.h[1] = *(const v8bf*)(p + 8);
  return u.v;
}

// ---------------------------------------------------------------------------
// Tiled GEMM: Out[M,N] = A[M,K] * W[K,N] (+bias). 128x128x64 tiles, 256 thr,
// 8 waves in a 4x2 grid, 2x4 16x16 accumulators each, 16 WMMA per K-step.
// ---------------------------------------------------------------------------
template <bool A_BF16, bool OUT_BF16>
__global__ __launch_bounds__(256)
void gemm_wmma_kernel(const void* __restrict__ Aptr,
                      const float* __restrict__ W,
                      const float* __restrict__ bias,
                      void* __restrict__ Out, int M, int N, int K) {
  __shared__ __align__(16) __bf16 Al[128 * 64];   // [m][k]
  __shared__ __align__(16) __bf16 Bl[128 * 64];   // [n][k] (transposed)

  const int tid  = threadIdx.x;
  const int lane = tid & 31;
  const int wave = tid >> 5;
  const int wm   = wave >> 1;        // 0..3 -> 32 rows each
  const int wn   = wave & 1;         // 0..1 -> 64 cols each
  const int m0 = blockIdx.x * 128;
  const int n0 = blockIdx.y * 128;

  v8f acc[2][4];
  #pragma unroll
  for (int i = 0; i < 2; ++i)
    #pragma unroll
    for (int j = 0; j < 4; ++j)
      acc[i][j] = vzero();

  for (int k0 = 0; k0 < K; k0 += 64) {
    // ---- stage A tile (128x64) ----
    if (A_BF16) {
      const __bf16* A = (const __bf16*)Aptr;
      #pragma unroll
      for (int i = 0; i < 4; ++i) {
        int idx = tid + i * 256;             // 1024 x b128 (async when avail)
        int row = idx >> 3, c8 = (idx & 7) * 8;
        stage_b128(&Al[row * 64 + c8], A + (size_t)(m0 + row) * K + k0 + c8);
      }
    } else {
      const float* A = (const float*)Aptr;
      #pragma unroll
      for (int i = 0; i < 8; ++i) {
        int idx = tid + i * 256;             // 2048 x float4 (convert to bf16)
        int row = idx >> 4, c4 = (idx & 15) * 4;
        float4 f = *(const float4*)(A + (size_t)(m0 + row) * K + k0 + c4);
        __bf16* d = &Al[row * 64 + c4];
        d[0] = (__bf16)f.x; d[1] = (__bf16)f.y;
        d[2] = (__bf16)f.z; d[3] = (__bf16)f.w;
      }
    }
    // ---- stage B tile (64x128 -> [n][k], convert to bf16) ----
    #pragma unroll
    for (int i = 0; i < 8; ++i) {
      int idx = tid + i * 256;               // 2048 x float4
      int kr = idx >> 5, n4 = (idx & 31) * 4;
      float4 f = *(const float4*)(W + (size_t)(k0 + kr) * N + n0 + n4);
      Bl[(n4 + 0) * 64 + kr] = (__bf16)f.x;
      Bl[(n4 + 1) * 64 + kr] = (__bf16)f.y;
      Bl[(n4 + 2) * 64 + kr] = (__bf16)f.z;
      Bl[(n4 + 3) * 64 + kr] = (__bf16)f.w;
    }
    stage_wait();
    __syncthreads();

    #pragma unroll
    for (int kc = 0; kc < 2; ++kc) {
      v16bf a[2], b[4];
      #pragma unroll
      for (int mt = 0; mt < 2; ++mt)
        a[mt] = load_a_frag(Al, 64, wm * 32 + mt * 16, kc * 32, lane);
      #pragma unroll
      for (int nt = 0; nt < 4; ++nt)
        b[nt] = load_b_frag(Bl, 64, wn * 64 + nt * 16, kc * 32, lane);
      #pragma unroll
      for (int mt = 0; mt < 2; ++mt)
        #pragma unroll
        for (int nt = 0; nt < 4; ++nt)
          acc[mt][nt] = wmma_bf16(a[mt], b[nt], acc[mt][nt]);
    }
    __syncthreads();
  }

  // ---- store (C/D layout: M = e + 8*(lane>=16), N = lane&15) ----
  const int half = lane >> 4, nloc = lane & 15;
  #pragma unroll
  for (int mt = 0; mt < 2; ++mt)
    #pragma unroll
    for (int nt = 0; nt < 4; ++nt)
      #pragma unroll
      for (int e = 0; e < 8; ++e) {
        int row = m0 + wm * 32 + mt * 16 + e + half * 8;
        int col = n0 + wn * 64 + nt * 16 + nloc;
        float v = acc[mt][nt][e];
        if (OUT_BF16) {
          ((__bf16*)Out)[(size_t)row * N + col] = (__bf16)v;
        } else {
          float bo = bias ? bias[col] : 0.f;
          ((float*)Out)[(size_t)row * N + col] = v + bo;
        }
      }
}

// ---------------------------------------------------------------------------
// Retention: per (b,h), for 128 q-rows: stream 32-key blocks.
// S = Q K^T (WMMA) -> decay/scale/mask in f32 accs -> bf16 via LDS -> O += S V.
// Q/K tiles staged with async-to-LDS; next K block prefetched.
// ---------------------------------------------------------------------------
__global__ __launch_bounds__(256)
void retention_kernel(const __bf16* __restrict__ qb,
                      const __bf16* __restrict__ kb,
                      const __bf16* __restrict__ vb,
                      float* __restrict__ attn,
                      float* __restrict__ stats) {
  __shared__ __align__(16) __bf16 Ql[128 * 64];   // [qrow][d]
  __shared__ __align__(16) __bf16 Kl[32 * 64];    // [key][d]
  __shared__ __align__(16) __bf16 Vl[64 * 32];    // [d][key] transposed
  __shared__ __align__(16) __bf16 Sl[8 * 16 * 32];// per-wave S tile

  const int tid  = threadIdx.x;
  const int lane = tid & 31;
  const int wm   = tid >> 5;                  // 8 waves x 16 q-rows = 128
  const int qt   = blockIdx.x;                // q-tile of 128 rows
  const int b    = blockIdx.y / Hn;
  const int h    = blockIdx.y % Hn;
  const int half = lane >> 4, nloc = lane & 15;

  const float gamma = 1.0f - exp2f(-5.0f - (float)h);
  const float lg    = log2f(gamma);
  const float scale = 0.125f;                 // 1/sqrt(64)

  // stage Q tile (128 x 64 bf16) via async-to-LDS
  const size_t qbase = (size_t)(b * Tn + qt * 128) * Cn + h * Dhn;
  #pragma unroll
  for (int i = 0; i < 4; ++i) {
    int idx = tid + i * 256;                  // 1024 x b128
    int row = idx >> 3, c8 = (idx & 7) * 8;
    stage_b128(&Ql[row * 64 + c8], qb + qbase + (size_t)row * Cn + c8);
  }

  v8f o[4];
  #pragma unroll
  for (int nt = 0; nt < 4; ++nt) o[nt] = vzero();

  const int njb = qt * 4 + 4;                 // causal: keys up to this q-tile
  for (int jb = 0; jb < njb; ++jb) {
    __syncthreads();
    const size_t kvbase = (size_t)(b * Tn + jb * 32) * Cn + h * Dhn;
    {  // K tile: 32x64, one async b128 per thread
      int row = tid >> 3, c8 = (tid & 7) * 8;
      stage_b128(&Kl[row * 64 + c8], kb + kvbase + (size_t)row * Cn + c8);
    }
    {  // V tile transposed into [d][key] (register transpose)
      int key = tid >> 3, db = (tid & 7) * 8;
      U4U u; u.u = *(const uint4*)(vb + kvbase + (size_t)key * Cn + db);
      #pragma unroll
      for (int j = 0; j < 8; ++j) Vl[(db + j) * 32 + key] = u.b[j];
    }
    if (jb + 1 < njb)  // global_prefetch_b8 of next key block
      __builtin_prefetch(kb + kvbase + (size_t)32 * Cn + (tid & 7) * 8, 0, 1);
    stage_wait();
    __syncthreads();

    // ---- S = Q K^T : 16x32 per wave ----
    v8f s[2] = {vzero(), vzero()};
    #pragma unroll
    for (int kc = 0; kc < 2; ++kc) {          // contract Dh=64 in 2 chunks
      v16bf aq  = load_a_frag(Ql, 64, wm * 16, kc * 32, lane);
      v16bf bk0 = load_b_frag(Kl, 64, 0,  kc * 32, lane);
      v16bf bk1 = load_b_frag(Kl, 64, 16, kc * 32, lane);
      s[0] = wmma_bf16(aq, bk0, s[0]);
      s[1] = wmma_bf16(aq, bk1, s[1]);
    }
    // ---- decay * mask * scale, convert to bf16 A-layout via LDS ----
    #pragma unroll
    for (int nt = 0; nt < 2; ++nt)
      #pragma unroll
      for (int e = 0; e < 8; ++e) {
        int Mrow = e + half * 8;
        int qi = qt * 128 + wm * 16 + Mrow;
        int kj = jb * 32 + nt * 16 + nloc;
        float d = (float)(qi - kj);
        float w = (qi >= kj) ? scale * exp2f(d * lg) : 0.0f;
        Sl[wm * 512 + Mrow * 32 + nt * 16 + nloc] = (__bf16)(s[nt][e] * w);
      }
    __syncthreads();

    // ---- O += S V : 16x64 per wave ----
    v16bf aS = load_a_frag(&Sl[wm * 512], 32, 0, 0, lane);
    #pragma unroll
    for (int nt = 0; nt < 4; ++nt) {
      v16bf bv = load_b_frag(Vl, 32, nt * 16, 0, lane);
      o[nt] = wmma_bf16(aS, bv, o[nt]);
    }
  }

  // ---- write O (f32) + GroupNorm partial stats ----
  float sum = 0.f, ssq = 0.f;
  #pragma unroll
  for (int nt = 0; nt < 4; ++nt)
    #pragma unroll
    for (int e = 0; e < 8; ++e) {
      int Mrow = e + half * 8;
      size_t row = (size_t)(b * Tn + qt * 128 + wm * 16 + Mrow);
      int col = h * Dhn + nt * 16 + nloc;
      float v = o[nt][e];
      attn[row * Cn + col] = v;
      sum += v; ssq += v * v;
    }
  atomicAdd(&stats[(b * Hn + h) * 2 + 0], sum);
  atomicAdd(&stats[(b * Hn + h) * 2 + 1], ssq);
}

// ---------------------------------------------------------------------------
__global__ void zero_stats_kernel(float* stats) { stats[threadIdx.x] = 0.f; }

__global__ __launch_bounds__(256)
void groupnorm_kernel(const float* __restrict__ attn,
                      const float* __restrict__ stats,
                      const float* __restrict__ gn_w,
                      const float* __restrict__ gn_b,
                      __bf16* __restrict__ normed) {
  const int idx4 = blockIdx.x * 256 + threadIdx.x;
  const size_t base = (size_t)idx4 * 4;
  const int row = (int)(base / Cn), col = (int)(base % Cn);
  const int bh  = (row / Tn) * Hn + col / Dhn;
  const float invN = 1.0f / (float)(Tn * Dhn);
  float mean = stats[bh * 2 + 0] * invN;
  float var  = stats[bh * 2 + 1] * invN - mean * mean;
  float inv  = rsqrtf(var + EPSf);
  float4 x = *(const float4*)(attn + base);
  union { __bf16 b[4]; uint2 u; } p;
  p.b[0] = (__bf16)((x.x - mean) * inv * gn_w[col + 0] + gn_b[col + 0]);
  p.b[1] = (__bf16)((x.y - mean) * inv * gn_w[col + 1] + gn_b[col + 1]);
  p.b[2] = (__bf16)((x.z - mean) * inv * gn_w[col + 2] + gn_b[col + 2]);
  p.b[3] = (__bf16)((x.w - mean) * inv * gn_w[col + 3] + gn_b[col + 3]);
  *(uint2*)(normed + base) = p.u;
}

// ---------------------------------------------------------------------------
extern "C" void kernel_launch(void* const* d_in, const int* in_sizes, int n_in,
                              void* d_out, int out_size, void* d_ws, size_t ws_size,
                              hipStream_t stream) {
  const float* x    = (const float*)d_in[0];
  const float* wq   = (const float*)d_in[1];
  const float* wk   = (const float*)d_in[2];
  const float* wv   = (const float*)d_in[3];
  const float* wo   = (const float*)d_in[4];
  const float* bo   = (const float*)d_in[5];
  const float* gn_w = (const float*)d_in[6];
  const float* gn_b = (const float*)d_in[7];
  float* out = (float*)d_out;

  // workspace layout
  char* ws = (char*)d_ws;
  const size_t bf_mat = (size_t)Mn * Cn * sizeof(__bf16);           // 8 MB
  __bf16* qb   = (__bf16*)(ws);
  __bf16* kb   = (__bf16*)(ws + bf_mat);
  __bf16* vb   = (__bf16*)(ws + 2 * bf_mat);
  float* attn  = (float*)(ws + 3 * bf_mat);                         // 16 MB
  __bf16* nm   = (__bf16*)(ws + 3 * bf_mat + (size_t)Mn * Cn * 4);
  float* stats = (float*)(ws + 4 * bf_mat + (size_t)Mn * Cn * 4);

  dim3 ggrid(Mn / 128, Cn / 128);   // 32 x 8
  // 1) QKV projections (f32 in -> bf16 out, WMMA bf16)
  gemm_wmma_kernel<false, true><<<ggrid, 256, 0, stream>>>(x, wq, nullptr, qb, Mn, Cn, Cn);
  gemm_wmma_kernel<false, true><<<ggrid, 256, 0, stream>>>(x, wk, nullptr, kb, Mn, Cn, Cn);
  gemm_wmma_kernel<false, true><<<ggrid, 256, 0, stream>>>(x, wv, nullptr, vb, Mn, Cn, Cn);

  // 2) retention + stats
  zero_stats_kernel<<<1, Bn * Hn * 2, 0, stream>>>(stats);
  retention_kernel<<<dim3(Tn / 128, Bn * Hn), 256, 0, stream>>>(qb, kb, vb, attn, stats);

  // 3) GroupNorm -> bf16
  groupnorm_kernel<<<((size_t)Mn * Cn / 4) / 256, 256, 0, stream>>>(attn, stats, gn_w, gn_b, nm);

  // 4) output projection (bf16 in -> f32 out + bias)
  gemm_wmma_kernel<true, false><<<ggrid, 256, 0, stream>>>(nm, wo, bo, out, Mn, Cn, Cn);
}
